// SumDecoder_44830868635753
// MI455X (gfx1250) — matrix operand order repaired and checked by reading, not compiled, not run on hardware
//
#include <hip/hip_runtime.h>
#include <hip/hip_bf16.h>
#include <math.h>
#include <stdint.h>

#define B_  64
#define E_  512
#define H_  1024
#define V_  50257
#define KV_ 1000
#define L_  512

typedef __attribute__((ext_vector_type(16))) _Float16 v16h;
typedef __attribute__((ext_vector_type(8)))  _Float16 v8h;
typedef __attribute__((ext_vector_type(8)))  float    v8f;

union AFrag { v16h v; _Float16 h[16]; };
union CFrag { v8f  v; float    f[8];  };
union H8    { v8h  v; _Float16 h[8];  };
union F4    { float4 v; float  f[4];  };

// ---------------------------------------------------------------------------
// Tiled WMMA GEMM, software-pipelined with double-buffered LDS.
//   C[M,N] = epi(A[M,K] @ Bw[K,N])
//   - K must be a multiple of 32 (true for every call site here)
//   - GM/GN: enable row/col bounds guards (only kproj / vocab GEMM need them)
//   - A16: A is f16; ASYNCA additionally streams the A tile into LDS with
//     per-lane GLOBAL_LOAD_ASYNC_TO_LDS_B128 (ASYNCcnt), no VGPR staging.
//   - Bw is f32 weights, converted to f16 into LDS transposed [n][k] so
//     fragment reads are contiguous ds_load_b128.
//   - epilogue: optional accumulate, col bias, rowgroup bias rowb[(row>>rsh)*N+col],
//     shifted add shv[row*sh_w + col-sh_off] for col>=sh_off, tanh, f16/f32 store
// ---------------------------------------------------------------------------
template<bool A16, bool TANH, bool ACC, bool OUT16, bool BIASN, bool ROWB,
         bool SHIFT, bool GM, bool GN, bool ASYNCA>
__global__ __launch_bounds__(256) void gemm_wmma(
    const void* __restrict__ Av, const float* __restrict__ Bw,
    float* __restrict__ Cf, _Float16* __restrict__ Ch,
    const float* __restrict__ bias,
    const float* __restrict__ rowb, int rsh,
    const float* __restrict__ shv, int sh_off, int sh_w,
    int M, int N, int K)
{
  static_assert(!ASYNCA || (A16 && !GM), "async A path requires f16 A, no row guard");
  constexpr int BM = 64, BN = 64, BK = 32;
  constexpr int LDS_A = 40, LDS_B = 40;              // half strides, rows 16B aligned
  __shared__ alignas(16) _Float16 As[2][BM * LDS_A]; // [row][k]
  __shared__ alignas(16) _Float16 Bs[2][BN * LDS_B]; // [col][k] (transposed)

  const int tid  = threadIdx.x;
  const int lane = tid & 31;
  const int wave = tid >> 5;
  const int wm   = wave & 3;                         // 4 row-tiles of 16
  const int wn   = wave >> 2;                        // 2 col-tiles of 32
  const int rowBase = blockIdx.y * BM;
  const int colBase = blockIdx.x * BN;

  const _Float16* A_h = (const _Float16*)Av;
  const float*    A_f = (const float*)Av;

  // per-thread staging chunk (8 consecutive elements)
  const int ar = tid >> 2, ac = (tid & 3) * 8;       // A: row, col
  const int bk = tid >> 3, bc = (tid & 7) * 8;       // B: k-row, col
  const int agr = rowBase + ar;
  const int bgc = colBase + bc;

  float ra[8], rb[8];

  auto loadA_regs = [&](int k0) {
    if (!GM || agr < M) {
      const size_t abase = (size_t)agr * K + k0 + ac;
      if constexpr (A16) {
        H8 t; t.v = *(const v8h*)(A_h + abase);
#pragma unroll
        for (int j = 0; j < 8; ++j) ra[j] = (float)t.h[j];
      } else {
        F4 t0, t1;
        t0.v = *(const float4*)(A_f + abase);
        t1.v = *(const float4*)(A_f + abase + 4);
#pragma unroll
        for (int j = 0; j < 4; ++j) { ra[j] = t0.f[j]; ra[4 + j] = t1.f[j]; }
      }
    } else {
#pragma unroll
      for (int j = 0; j < 8; ++j) ra[j] = 0.f;
    }
  };

  // per-lane async copy: 16B global f16 chunk -> LDS, tracked by ASYNCcnt
  auto issueA_async = [&](int k0, int bufn) {
    const _Float16* gsrc = A_h + (size_t)agr * K + k0 + ac;
    unsigned loff = (unsigned)(uintptr_t)(&As[bufn][ar * LDS_A + ac]); // LDS addr = flat[31:0]
    asm volatile("global_load_async_to_lds_b128 %0, %1, off"
                 :: "v"(loff), "v"(gsrc) : "memory");
  };

  auto loadB = [&](int k0) {
    const size_t bbase = (size_t)(k0 + bk) * N + bgc;
    if constexpr (!GN) {                             // N multiple of 8: b128 loads
      F4 t0, t1;
      t0.v = *(const float4*)(Bw + bbase);
      t1.v = *(const float4*)(Bw + bbase + 4);
#pragma unroll
      for (int j = 0; j < 4; ++j) { rb[j] = t0.f[j]; rb[4 + j] = t1.f[j]; }
    } else {                                         // odd N (vocab): guarded scalars
#pragma unroll
      for (int j = 0; j < 8; ++j) rb[j] = (bgc + j < N) ? Bw[bbase + j] : 0.f;
    }
  };

  auto storeA = [&](int bufn) {
    H8 s;
#pragma unroll
    for (int j = 0; j < 8; ++j) s.h[j] = (_Float16)ra[j];
    *(v8h*)(&As[bufn][ar * LDS_A + ac]) = s.v;       // one ds_store_b128
  };

  auto storeB = [&](int bufn) {
#pragma unroll
    for (int j = 0; j < 8; ++j)                      // transposed scatter
      Bs[bufn][(bc + j) * LDS_B + bk] = (_Float16)rb[j];
  };

  v8f acc0 = {}, acc1 = {};

  // ---- prologue: tile 0 ----
  if constexpr (ASYNCA) issueA_async(0, 0); else loadA_regs(0);
  loadB(0);
  if constexpr (!ASYNCA) storeA(0);
  storeB(0);
  if constexpr (ASYNCA) asm volatile("s_wait_asynccnt 0x0" ::: "memory");
  __syncthreads();

  int buf = 0;
  const int nk = K / BK;
  for (int i = 0; i < nk; ++i) {
    const bool hasNext = (i + 1 < nk);
    if (hasNext) {                                   // overlaps with WMMAs below
      if constexpr (ASYNCA) issueA_async((i + 1) * BK, buf ^ 1);
      else                  loadA_regs((i + 1) * BK);
      loadB((i + 1) * BK);
      if (i + 2 < nk)                                // stream tile i+2 (global_prefetch_b8)
        __builtin_prefetch(&Bw[(size_t)((i + 2) * BK + bk) * N + bgc], 0, 1);
    }

    // A fragment, contiguous K order: lanes 0-15 K base 0, lanes 16-31 base 16;
    // halfs 0..15 -> K=base+0..15  (two b128 LDS loads)
    AFrag af, bf0, bf1;
    const _Float16* ap = &As[buf][(wm * 16 + (lane & 15)) * LDS_A + ((lane & 16) ? 16 : 0)];
#pragma unroll
    for (int j = 0; j < 8; ++j) { af.h[j] = ap[j]; af.h[8 + j] = ap[8 + j]; }

    // B fragment: col = lane%16, K = (lane<16?0:16)+hi -> contiguous in Bs[col][k]
    const int kb = (lane & 16) ? 16 : 0;
    const _Float16* bp0 = &Bs[buf][(wn * 32 + (lane & 15)) * LDS_B + kb];
    const _Float16* bp1 = bp0 + 16 * LDS_B;
#pragma unroll
    for (int j = 0; j < 16; ++j) { bf0.h[j] = bp0[j]; bf1.h[j] = bp1[j]; }

    acc0 = __builtin_amdgcn_wmma_f32_16x16x32_f16(false, af.v, false, bf0.v,
                                                  (short)0, acc0, false, false);
    acc1 = __builtin_amdgcn_wmma_f32_16x16x32_f16(false, af.v, false, bf1.v,
                                                  (short)0, acc1, false, false);

    if (hasNext) {
      if constexpr (!ASYNCA) storeA(buf ^ 1);        // other buffer: no race with readers
      storeB(buf ^ 1);
      if constexpr (ASYNCA) asm volatile("s_wait_asynccnt 0x0" ::: "memory");
      __syncthreads();                               // single barrier per K-step
      buf ^= 1;
    }
  }

  // C/D layout: VGPR r -> M = r + 8*(lane>=16), N = lane%16
  CFrag c0, c1; c0.v = acc0; c1.v = acc1;
  const int rloc = wm * 16 + ((lane & 16) ? 8 : 0);
  const int nloc = wn * 32 + (lane & 15);
#pragma unroll
  for (int r = 0; r < 8; ++r) {
    int grow = rowBase + rloc + r;
    if (GM && grow >= M) continue;
#pragma unroll
    for (int half = 0; half < 2; ++half) {
      int gcol = colBase + nloc + half * 16;
      if (GN && gcol >= N) continue;
      float v = half ? c1.f[r] : c0.f[r];
      size_t oi = (size_t)grow * N + gcol;
      if constexpr (ACC)   v += Cf[oi];
      if constexpr (BIASN) v += bias[gcol];
      if constexpr (ROWB)  v += rowb[(size_t)(grow >> rsh) * N + gcol];
      if constexpr (SHIFT) { if (gcol >= sh_off) v += shv[(size_t)grow * sh_w + (gcol - sh_off)]; }
      if constexpr (TANH)  v = tanhf(v);
      if constexpr (OUT16) Ch[oi] = (_Float16)v;
      else                 Cf[oi] = v;
    }
  }
}

// ---------------------------------------------------------------------------
// Small kernels
// ---------------------------------------------------------------------------
__global__ void embed_k(const int* __restrict__ ids, const float* __restrict__ emb,
                        float* __restrict__ x) {
  int i = blockIdx.x * 256 + threadIdx.x;
  if (i >= B_ * E_) return;
  int b = i / E_, e = i % E_;
  x[i] = emb[(size_t)ids[b] * E_ + e];
}

__global__ void lstm_k(const float* __restrict__ gates, const float* __restrict__ c_in,
                       float* __restrict__ h_t, float* __restrict__ c_t) {
  int i = blockIdx.x * 256 + threadIdx.x;
  if (i >= B_ * H_) return;
  int b = i / H_, j = i % H_;
  const float* g = gates + (size_t)b * 4 * H_;
  float ig = 1.f / (1.f + __expf(-g[j]));
  float fg = 1.f / (1.f + __expf(-g[H_ + j]));
  float gg = tanhf(g[2 * H_ + j]);
  float og = 1.f / (1.f + __expf(-g[3 * H_ + j]));
  float c  = fg * c_in[i] + ig * gg;
  c_t[i] = c;
  h_t[i] = og * tanhf(c);
}

// wave-per-row dot with f16 A and tiny f32 weight (N==1 layers)
__global__ void gemv_k(const _Float16* __restrict__ A, const float* __restrict__ w,
                       const float* __restrict__ bias, float* __restrict__ out,
                       int M, int K) {
  int row  = blockIdx.x * (blockDim.x >> 5) + (threadIdx.x >> 5);
  int lane = threadIdx.x & 31;
  if (row >= M) return;
  const _Float16* ar = A + (size_t)row * K;
  float s = 0.f;
  for (int k = lane; k < K; k += 32) s += (float)ar[k] * w[k];
  for (int off = 16; off; off >>= 1) s += __shfl_xor(s, off, 32);
  if (lane == 0) out[row] = s + bias[0];
}

__global__ void softmax_k(const float* __restrict__ x, float* __restrict__ y, int n) {
  int b = blockIdx.x, t = threadIdx.x;
  __shared__ float red[256];
  const float* xr = x + (size_t)b * n;
  float m = -INFINITY;
  for (int i = t; i < n; i += 256) m = fmaxf(m, xr[i]);
  red[t] = m; __syncthreads();
  for (int s = 128; s; s >>= 1) { if (t < s) red[t] = fmaxf(red[t], red[t + s]); __syncthreads(); }
  m = red[0]; __syncthreads();
  float sum = 0.f;
  for (int i = t; i < n; i += 256) sum += __expf(xr[i] - m);
  red[t] = sum; __syncthreads();
  for (int s = 128; s; s >>= 1) { if (t < s) red[t] += red[t + s]; __syncthreads(); }
  float inv = 1.f / red[0];
  for (int i = t; i < n; i += 256) y[(size_t)b * n + i] = __expf(xr[i] - m) * inv;
}

__global__ void wsum_k(const float* __restrict__ at, const float* __restrict__ enc,
                       float* __restrict__ hctx) {
  int h = blockIdx.x * 256 + threadIdx.x;
  int b = blockIdx.y;
  if (h >= H_) return;
  const float* e = enc + (size_t)b * L_ * H_ + h;
  const float* a = at + (size_t)b * L_;
  float s = 0.f;
  for (int l = 0; l < L_; ++l) s += a[l] * e[(size_t)l * H_];
  hctx[(size_t)b * H_ + h] = s;
}

// t1k[b,kv,:] = tanh(kbias[b,:] + kproj[kv,:])  (factored key-attention layer 1)
__global__ void t1k_k(const float* __restrict__ kbias, const float* __restrict__ kproj,
                      _Float16* __restrict__ t1k) {
  size_t i = (size_t)blockIdx.x * 256 + threadIdx.x;
  if (i >= (size_t)B_ * KV_ * E_) return;
  int    col = (int)(i % E_);
  size_t row = i / E_;
  int b  = (int)(row / KV_);
  int kv = (int)(row % KV_);
  t1k[i] = (_Float16)tanhf(kbias[(size_t)b * E_ + col] + kproj[(size_t)kv * E_ + col]);
}

__global__ void cat_k(const float* __restrict__ h_t, const float* __restrict__ hctx,
                      float* __restrict__ catb) {
  int i = blockIdx.x * 256 + threadIdx.x;
  if (i >= B_ * 2 * H_) return;
  int b = i / (2 * H_), j = i % (2 * H_);
  catb[i] = (j < H_) ? h_t[(size_t)b * H_ + j] : hctx[(size_t)b * H_ + (j - H_)];
}

__global__ void logsoftmax_k(float* __restrict__ y, int n) {
  int b = blockIdx.x, t = threadIdx.x;
  __shared__ float red[256];
  float* yr = y + (size_t)b * n;
  float m = -INFINITY;
  for (int i = t; i < n; i += 256) m = fmaxf(m, yr[i]);
  red[t] = m; __syncthreads();
  for (int s = 128; s; s >>= 1) { if (t < s) red[t] = fmaxf(red[t], red[t + s]); __syncthreads(); }
  m = red[0]; __syncthreads();
  float sum = 0.f;
  for (int i = t; i < n; i += 256) sum += __expf(yr[i] - m);
  red[t] = sum; __syncthreads();
  for (int s = 128; s; s >>= 1) { if (t < s) red[t] += red[t + s]; __syncthreads(); }
  float lse = m + __logf(red[0]);
  for (int i = t; i < n; i += 256) yr[i] = yr[i] - lse;
}

// ---------------------------------------------------------------------------
extern "C" void kernel_launch(void* const* d_in, const int* in_sizes, int n_in,
                              void* d_out, int out_size, void* d_ws, size_t ws_size,
                              hipStream_t stream) {
  (void)in_sizes; (void)n_in; (void)out_size; (void)ws_size;
  const int*   ids  = (const int*)  d_in[0];
  const float* h_in = (const float*)d_in[1];
  const float* c_in = (const float*)d_in[2];
  const float* kmat = (const float*)d_in[3];
  const float* ctxv = (const float*)d_in[4];
  const float* enc  = (const float*)d_in[5];
  const float* emb  = (const float*)d_in[6];
  const float* W_ih = (const float*)d_in[7];
  const float* W_hh = (const float*)d_in[8];
  const float* b_ih = (const float*)d_in[9];
  const float* b_hh = (const float*)d_in[10];
  const float* aW1  = (const float*)d_in[11];
  const float* ab1  = (const float*)d_in[12];
  const float* aW2  = (const float*)d_in[13];
  const float* ab2  = (const float*)d_in[14];
  const float* aW3  = (const float*)d_in[15];
  const float* ab3  = (const float*)d_in[16];
  const float* kW1  = (const float*)d_in[17];
  const float* kb1  = (const float*)d_in[18];
  const float* kW2  = (const float*)d_in[19];
  const float* kb2  = (const float*)d_in[20];
  const float* kW3  = (const float*)d_in[21];
  const float* kb3  = (const float*)d_in[22];
  const float* alW  = (const float*)d_in[23];
  const float* alb  = (const float*)d_in[24];

  // outputs: y_t [B,V] | h_t [B,H] | c_t [B,H]
  float* y   = (float*)d_out;
  float* h_t = y + (size_t)B_ * V_;
  float* c_t = h_t + (size_t)B_ * H_;

  // workspace layout
  char*  ws  = (char*)d_ws;
  size_t off = 0;
  auto take = [&](size_t bytes) { char* p = ws + off; off += (bytes + 255) & ~(size_t)255; return p; };
  float* x      = (float*)take((size_t)B_ * E_ * 4);
  float* gates  = (float*)take((size_t)B_ * 4 * H_ * 4);
  float* hbiasA = (float*)take((size_t)B_ * E_ * 4);
  float* kbias  = (float*)take((size_t)B_ * E_ * 4);
  float* kproj  = (float*)take((size_t)KV_ * E_ * 4);
  float* scores = (float*)take((size_t)B_ * L_ * 4);
  float* at     = (float*)take((size_t)B_ * L_ * 4);
  float* hctx   = (float*)take((size_t)B_ * H_ * 4);
  float* ukv    = (float*)take((size_t)B_ * KV_ * 4);
  float* catb   = (float*)take((size_t)B_ * 2 * H_ * 4);
  char*  arena  = ws + off;                                   // big f16 region, reused
  _Float16* t1  = (_Float16*)arena;                                    // [B*L, E]
  _Float16* t2  = (_Float16*)(arena + (size_t)B_ * L_ * E_ * 2);       // [B*L, E]
  _Float16* t1k = (_Float16*)arena;                                    // [B*KV, E] (reuse)
  _Float16* t2k = (_Float16*)(arena + (size_t)B_ * KV_ * E_ * 2);      // [B*KV, E]

  auto g64 = [](int n) { return (n + 63) / 64; };
  dim3 blk(256);

  // 0) embedding gather
  embed_k<<<(B_ * E_ + 255) / 256, blk, 0, stream>>>(ids, emb, x);

  // 1) LSTM gates = x@W_ih + b_ih, then += h@W_hh + b_hh
  gemm_wmma<false,false,false,false,true,false,false,false,false,false>
      <<<dim3(g64(4 * H_), g64(B_)), blk, 0, stream>>>(
      x, W_ih, gates, nullptr, b_ih, nullptr, 0, nullptr, 0, 0, B_, 4 * H_, E_);
  gemm_wmma<false,false,true,false,true,false,false,false,false,false>
      <<<dim3(g64(4 * H_), g64(B_)), blk, 0, stream>>>(
      h_in, W_hh, gates, nullptr, b_hh, nullptr, 0, nullptr, 0, 0, B_, 4 * H_, H_);
  lstm_k<<<(B_ * H_ + 255) / 256, blk, 0, stream>>>(gates, c_in, h_t, c_t);

  // 2) attention: hbiasA = h_t @ aW1[H:] + ab1 (factored per-batch bias)
  gemm_wmma<false,false,false,false,true,false,false,false,false,false>
      <<<dim3(g64(E_), 1), blk, 0, stream>>>(
      h_t, aW1 + (size_t)H_ * E_, hbiasA, nullptr, ab1, nullptr, 0, nullptr, 0, 0, B_, E_, H_);
  //    t1 = tanh(enc @ aW1[:H] + hbiasA[row>>9])   [B*L, E] f16
  gemm_wmma<false,true,false,true,false,true,false,false,false,false>
      <<<dim3(g64(E_), g64(B_ * L_)), blk, 0, stream>>>(
      enc, aW1, nullptr, t1, nullptr, hbiasA, 9, nullptr, 0, 0, B_ * L_, E_, H_);
  //    t2 = tanh(t1 @ aW2 + ab2)   (f16 A, async-to-LDS staging)
  gemm_wmma<true,true,false,true,true,false,false,false,false,true>
      <<<dim3(g64(E_), g64(B_ * L_)), blk, 0, stream>>>(
      t1, aW2, nullptr, t2, ab2, nullptr, 0, nullptr, 0, 0, B_ * L_, E_, E_);
  //    scores = t2 @ aW3 + ab3 ; softmax over L ; h_ctx
  gemv_k<<<(B_ * L_ + 7) / 8, blk, 0, stream>>>(t2, aW3, ab3, scores, B_ * L_, E_);
  softmax_k<<<B_, blk, 0, stream>>>(scores, at, L_);
  wsum_k<<<dim3((H_ + 255) / 256, B_), blk, 0, stream>>>(at, enc, hctx);

  // 3) key attention, factored layer 1
  gemm_wmma<false,false,false,false,true,false,false,false,false,false>
      <<<dim3(g64(E_), 1), blk, 0, stream>>>(
      ctxv, kW1, kbias, nullptr, kb1, nullptr, 0, nullptr, 0, 0, B_, E_, H_);
  gemm_wmma<false,false,true,false,false,false,false,false,false,false>
      <<<dim3(g64(E_), 1), blk, 0, stream>>>(
      h_t, kW1 + (size_t)H_ * E_, kbias, nullptr, nullptr, nullptr, 0, nullptr, 0, 0, B_, E_, H_);
  //    kproj = k @ kW1[2H:]   [KV, E]  (batch independent; M=1000 -> GM)
  gemm_wmma<false,false,false,false,false,false,false,true,false,false>
      <<<dim3(g64(E_), g64(KV_)), blk, 0, stream>>>(
      kmat, kW1 + (size_t)2 * H_ * E_, kproj, nullptr, nullptr, nullptr, 0, nullptr, 0, 0, KV_, E_, E_);
  //    t1k = tanh(kbias[b] + kproj[kv])   [B*KV, E] f16
  {
    size_t tot = (size_t)B_ * KV_ * E_;
    t1k_k<<<(unsigned)((tot + 255) / 256), blk, 0, stream>>>(kbias, kproj, t1k);
  }
  //    t2k = tanh(t1k @ kW2 + kb2)   (f16 A, async-to-LDS staging)
  gemm_wmma<true,true,false,true,true,false,false,false,false,true>
      <<<dim3(g64(E_), g64(B_ * KV_)), blk, 0, stream>>>(
      t1k, kW2, nullptr, t2k, kb2, nullptr, 0, nullptr, 0, 0, B_ * KV_, E_, E_);
  //    uk = t2k @ kW3 + kb3   [B, KV]
  gemv_k<<<(B_ * KV_ + 7) / 8, blk, 0, stream>>>(t2k, kW3, kb3, ukv, B_ * KV_, E_);

  // 4) vocab logits: o = cat(h_t, h_ctx) @ alW + alb (+ uk in last KV cols); N odd -> GN
  cat_k<<<(B_ * 2 * H_ + 255) / 256, blk, 0, stream>>>(h_t, hctx, catb);
  gemm_wmma<false,false,false,false,true,false,true,false,true,false>
      <<<dim3(g64(V_), g64(B_)), blk, 0, stream>>>(
      catb, alW, y, nullptr, alb, nullptr, 0, ukv, V_ - KV_, KV_, B_, V_, 2 * H_);

  // 5) log-softmax over V, in place in d_out
  logsoftmax_k<<<B_, blk, 0, stream>>>(y, V_);
}
